// AdaGNN_47665547051069
// MI455X (gfx1250) — matrix operand order repaired and checked by reading, not compiled, not run on hardware
//
#include <hip/hip_runtime.h>
#include <hip/hip_bf16.h>

#define N_NODES 50000
#define N_EDGES 800000
#define NFEAT 256
#define NHID 128
#define NCLASS 40
#define NCLASS_PAD 48

typedef __attribute__((ext_vector_type(16))) _Float16 v16h;
typedef __attribute__((ext_vector_type(8)))  float    v8f;

// ---- CDNA5 async global->LDS copy helpers (ASYNCcnt path) ----
// Each active lane copies 16 bytes: LDS[lds_off] = MEM[gaddr .. gaddr+15]
__device__ __forceinline__ void async_copy_b128(unsigned lds_off, const void* gptr) {
    asm volatile("global_load_async_to_lds_b128 %0, %1, off"
                 :: "v"(lds_off), "v"((unsigned long long)(uintptr_t)gptr)
                 : "memory");
}
__device__ __forceinline__ void wait_async0() {
    asm volatile("s_wait_asynccnt 0" ::: "memory");
}

// ---------------- utility: zero fill ----------------
__global__ void fill0_kernel(float* __restrict__ p, long n) {
    long i = (long)blockIdx.x * blockDim.x + threadIdx.x;
    if (i < n) p[i] = 0.0f;
}

// ---------------- SpMM scatter: one wave per edge ----------------
// y[rows[e], :] += vals[e] * x[cols[e], :]
__global__ void spmm_scatter_kernel(const int* __restrict__ rows,
                                    const int* __restrict__ cols,
                                    const float* __restrict__ vals,
                                    const float* __restrict__ x,
                                    float* __restrict__ y,
                                    int F, int E) {
    int wid  = (int)((blockIdx.x * (long)blockDim.x + threadIdx.x) >> 5);
    int lane = threadIdx.x & 31;
    if (wid >= E) return;
    // software-pipeline hint for the edge index stream (global_prefetch_b8)
    if (wid + 2048 < E) {
        __builtin_prefetch(cols + wid + 2048, 0, 0);
        __builtin_prefetch(rows + wid + 2048, 0, 0);
    }
    int r = rows[wid];
    int c = cols[wid];
    float v = vals[wid];
    const float* xs = x + (long)c * F;
    float*       yd = y + (long)r * F;
    for (int f = lane; f < F; f += 32) {
        unsafeAtomicAdd(&yd[f], v * xs[f]);  // global_atomic_add_f32
    }
}

// ---------------- elementwise builders ----------------
// a16 = f16(x - e * (diag[k]+1)), k = i & (F-1)
__global__ void build_a_kernel(const float* __restrict__ x,
                               const float* __restrict__ e,
                               const float* __restrict__ diag,
                               _Float16* __restrict__ a,
                               int fmask, long total) {
    long i = (long)blockIdx.x * blockDim.x + threadIdx.x;
    if (i >= total) return;
    int k = (int)(i & fmask);
    a[i] = (_Float16)(x[i] - e[i] * (diag[k] + 1.0f));
}

// h = relu(h - e * hd[k])   (hidden layers, in place)
__global__ void hidden_update_kernel(float* __restrict__ h,
                                     const float* __restrict__ e,
                                     const float* __restrict__ hd,
                                     long total) {
    long i = (long)blockIdx.x * blockDim.x + threadIdx.x;
    if (i >= total) return;
    int k = (int)(i & (NHID - 1));
    h[i] = fmaxf(h[i] - e[i] * hd[k], 0.0f);
}

// ---------------- weight transpose+convert ----------------
// W1: NFEAT x NHID (row major)  ->  Bt: NHID x NFEAT f16
__global__ void conv_w1t_kernel(const float* __restrict__ W, _Float16* __restrict__ Bt) {
    int i = blockIdx.x * blockDim.x + threadIdx.x;   // over NHID*NFEAT
    if (i >= NHID * NFEAT) return;
    int n = i >> 8;       // output column 0..127
    int k = i & 255;      // k 0..255
    Bt[i] = (_Float16)W[k * NHID + n];
}
// W2: NHID x NCLASS -> Bt: NCLASS_PAD x NHID f16 (rows >= NCLASS zero)
__global__ void conv_w2t_kernel(const float* __restrict__ W, _Float16* __restrict__ Bt) {
    int i = blockIdx.x * blockDim.x + threadIdx.x;   // over NCLASS_PAD*NHID
    if (i >= NCLASS_PAD * NHID) return;
    int n = i >> 7;       // 0..47
    int k = i & 127;
    Bt[i] = (_Float16)((n < NCLASS) ? W[k * NCLASS + n] : 0.0f);
}

// ---------------- WMMA GEMM 1: H = relu(A @ W1 + b1) ----------------
// A: N x 256 f16, Bt: 128 x 256 f16 (W1^T), H: N x 128 f32
// block = 256 threads = 8 waves; wave w -> N-tile w; block -> 16-row M-tile.
// A tile (16 rows x 256 = 8KB, contiguous) staged to LDS via async copy and
// shared by all 8 waves; B panels stream from global (L2 resident, per-wave unique).
__global__ void gemm1_wmma_kernel(const _Float16* __restrict__ A,
                                  const _Float16* __restrict__ Bt,
                                  const float* __restrict__ bias,
                                  float* __restrict__ H) {
    __shared__ _Float16 sA[16 * NFEAT];           // 8192 bytes
    const int tid   = threadIdx.x;
    const int lane  = tid & 31;
    const int wave  = tid >> 5;                   // 0..7
    const int mbase = blockIdx.x * 16;
    const int half  = lane >> 4;                  // 0 or 1
    const int mrow  = lane & 15;
    const int ncol  = wave * 16 + (lane & 15);

    // stage A tile: 8192 contiguous bytes, 16B per lane per issue, 2 issues
    {
        const char* gsrc = (const char*)(A + (size_t)mbase * NFEAT);
        unsigned lbase = (unsigned)(uintptr_t)(&sA[0]);
        #pragma unroll
        for (int p = 0; p < 2; ++p) {
            int off = (tid + p * 256) * 16;       // 0..8176
            async_copy_b128(lbase + off, gsrc + off);
        }
        wait_async0();
    }
    __syncthreads();

    const _Float16* brow = Bt + (long)ncol * NFEAT;

    v8f acc = {};
    #pragma unroll
    for (int kb = 0; kb < NFEAT; kb += 32) {
        union { v16h v; unsigned u[8]; } af, bf;
        #pragma unroll
        for (int t = 0; t < 4; ++t) {
            int ka = kb + half * 8 + 2 * t;       // A: VGPR t -> K=8*half+2t ; t+4 -> +16
            af.u[t]     = *(const unsigned*)&sA[mrow * NFEAT + ka];
            af.u[t + 4] = *(const unsigned*)&sA[mrow * NFEAT + ka + 16];
        }
        #pragma unroll
        for (int t = 0; t < 8; ++t) {             // B: VGPR t -> K=8*half+2t ; t>=4 -> +8
            int kk = kb + half * 8 + 2 * t + (t >= 4 ? 8 : 0);
            bf.u[t] = *(const unsigned*)(brow + kk);
        }
        acc = __builtin_amdgcn_wmma_f32_16x16x32_f16(
            false, af.v, false, bf.v, (short)0, acc, false, false);
    }
    float bv = bias[ncol];
    #pragma unroll
    for (int r = 0; r < 8; ++r) {
        int row = mbase + r + 8 * half;           // C/D: VGPR r -> M=r (+8 for hi half)
        H[(long)row * NHID + ncol] = fmaxf(acc[r] + bv, 0.0f);
    }
}

// ---------------- WMMA GEMM 2: out = relu(A2 @ W2 + b2) -> d_out (N x 40) ----------------
// A2: N x 128 f16, Bt: 48 x 128 f16, block = 96 threads = 3 waves (N-tiles 0..2)
__global__ void gemm2_wmma_kernel(const _Float16* __restrict__ A,
                                  const _Float16* __restrict__ Bt,
                                  const float* __restrict__ bias,
                                  float* __restrict__ out) {
    __shared__ _Float16 sA[16 * NHID];            // 4096 bytes
    const int tid   = threadIdx.x;
    const int lane  = tid & 31;
    const int wave  = tid >> 5;                   // 0..2
    const int mbase = blockIdx.x * 16;
    const int half  = lane >> 4;
    const int mrow  = lane & 15;
    const int ncol  = wave * 16 + (lane & 15);    // 0..47

    // stage A tile: 4096 contiguous bytes, 16B per lane per issue
    {
        const char* gsrc = (const char*)(A + (size_t)mbase * NHID);
        unsigned lbase = (unsigned)(uintptr_t)(&sA[0]);
        for (int off = tid * 16; off < 16 * NHID * 2; off += 96 * 16) {
            async_copy_b128(lbase + off, gsrc + off);
        }
        wait_async0();
    }
    __syncthreads();

    const _Float16* brow = Bt + (long)ncol * NHID;

    v8f acc = {};
    #pragma unroll
    for (int kb = 0; kb < NHID; kb += 32) {
        union { v16h v; unsigned u[8]; } af, bf;
        #pragma unroll
        for (int t = 0; t < 4; ++t) {
            int ka = kb + half * 8 + 2 * t;
            af.u[t]     = *(const unsigned*)&sA[mrow * NHID + ka];
            af.u[t + 4] = *(const unsigned*)&sA[mrow * NHID + ka + 16];
        }
        #pragma unroll
        for (int t = 0; t < 8; ++t) {
            int kk = kb + half * 8 + 2 * t + (t >= 4 ? 8 : 0);
            bf.u[t] = *(const unsigned*)(brow + kk);
        }
        acc = __builtin_amdgcn_wmma_f32_16x16x32_f16(
            false, af.v, false, bf.v, (short)0, acc, false, false);
    }
    if (ncol < NCLASS) {
        float bv = bias[ncol];
        #pragma unroll
        for (int r = 0; r < 8; ++r) {
            int row = mbase + r + 8 * half;
            out[(long)row * NCLASS + ncol] = fmaxf(acc[r] + bv, 0.0f);
        }
    }
}

// ---------------- log-softmax over 40 classes, one wave per row, in place ----------------
__global__ void log_softmax40_kernel(float* __restrict__ out, int nrows) {
    int wid  = threadIdx.x >> 5;
    int row  = blockIdx.x * (blockDim.x >> 5) + wid;
    if (row >= nrows) return;
    int lane = threadIdx.x & 31;
    float* p = out + (long)row * NCLASS;
    float a = p[lane];                                     // lane < 32 < 40
    bool  hasb = (lane + 32) < NCLASS;
    float b = hasb ? p[lane + 32] : -3.0e38f;
    float m = fmaxf(a, b);
    #pragma unroll
    for (int off = 16; off > 0; off >>= 1)
        m = fmaxf(m, __shfl_xor(m, off, 32));
    float s = __expf(a - m) + (hasb ? __expf(b - m) : 0.0f);
    #pragma unroll
    for (int off = 16; off > 0; off >>= 1)
        s += __shfl_xor(s, off, 32);
    float l = m + __logf(s);
    p[lane] = a - l;
    if (hasb) p[lane + 32] = b - l;
}

// ---------------- launcher ----------------
extern "C" void kernel_launch(void* const* d_in, const int* in_sizes, int n_in,
                              void* d_out, int out_size, void* d_ws, size_t ws_size,
                              hipStream_t stream) {
    const float* x      = (const float*)d_in[0];
    const int*   rows   = (const int*)  d_in[1];
    const int*   cols   = (const int*)  d_in[2];
    const float* vals   = (const float*)d_in[3];
    const float* diag1  = (const float*)d_in[4];
    const float* W1     = (const float*)d_in[5];
    const float* b1     = (const float*)d_in[6];
    const float* hdiags = (const float*)d_in[7];   // 2 x 128
    const float* diag2  = (const float*)d_in[8];
    const float* W2     = (const float*)d_in[9];
    const float* b2     = (const float*)d_in[10];
    float* out = (float*)d_out;

    char* ws = (char*)d_ws;
    float*     e1  = (float*)ws;                                  // N*256 f32
    _Float16*  a16 = (_Float16*)(ws + (size_t)N_NODES * NFEAT * 4);        // N*256 f16
    float*     h   = (float*)(ws + (size_t)N_NODES * NFEAT * 6);           // N*128 f32
    _Float16*  w1t = (_Float16*)(ws + (size_t)N_NODES * NFEAT * 6
                                    + (size_t)N_NODES * NHID * 4);         // 128*256 f16
    _Float16*  w2t = (_Float16*)((char*)w1t + (size_t)NHID * NFEAT * 2);   // 48*128 f16

    const long totF = (long)N_NODES * NFEAT;   // 12.8M
    const long totH = (long)N_NODES * NHID;    // 6.4M

    // weight conversion (tiny)
    conv_w1t_kernel<<<(NHID * NFEAT + 255) / 256, 256, 0, stream>>>(W1, w1t);
    conv_w2t_kernel<<<(NCLASS_PAD * NHID + 255) / 256, 256, 0, stream>>>(W2, w2t);

    // layer 1: e1 = L @ x ; A = x - e1*(diag1+1) ; h = relu(A @ W1 + b1)
    fill0_kernel<<<(int)((totF + 255) / 256), 256, 0, stream>>>(e1, totF);
    spmm_scatter_kernel<<<(N_EDGES * 32 + 255) / 256, 256, 0, stream>>>(
        rows, cols, vals, x, e1, NFEAT, N_EDGES);
    build_a_kernel<<<(int)((totF + 255) / 256), 256, 0, stream>>>(
        x, e1, diag1, a16, NFEAT - 1, totF);
    gemm1_wmma_kernel<<<N_NODES / 16, 256, 0, stream>>>(a16, w1t, b1, h);

    // hidden layers: h = relu(h - (L@h) * hd[l])
    for (int l = 0; l < 2; ++l) {
        fill0_kernel<<<(int)((totH + 255) / 256), 256, 0, stream>>>(e1, totH);
        spmm_scatter_kernel<<<(N_EDGES * 32 + 255) / 256, 256, 0, stream>>>(
            rows, cols, vals, h, e1, NHID, N_EDGES);
        hidden_update_kernel<<<(int)((totH + 255) / 256), 256, 0, stream>>>(
            h, e1, hdiags + l * NHID, totH);
    }

    // layer 2: e1 = L@h ; A2 = h - e1*(diag2+1) ; out = relu(A2 @ W2 + b2)
    fill0_kernel<<<(int)((totH + 255) / 256), 256, 0, stream>>>(e1, totH);
    spmm_scatter_kernel<<<(N_EDGES * 32 + 255) / 256, 256, 0, stream>>>(
        rows, cols, vals, h, e1, NHID, N_EDGES);
    build_a_kernel<<<(int)((totH + 255) / 256), 256, 0, stream>>>(
        h, e1, diag2, a16, NHID - 1, totH);
    gemm2_wmma_kernel<<<N_NODES / 16, 96, 0, stream>>>(a16, w2t, b2, out);

    // log-softmax in place on d_out
    log_softmax40_kernel<<<(N_NODES + 7) / 8, 256, 0, stream>>>(out, N_NODES);
}
